// FlexQMixer_9242769622516
// MI455X (gfx1250) — matrix operand order repaired and checked by reading, not compiled, not run on hardware
//
#include <hip/hip_runtime.h>

typedef __attribute__((ext_vector_type(16))) _Float16 v16h;
typedef __attribute__((ext_vector_type(8)))  _Float16 v8h;
typedef __attribute__((ext_vector_type(8)))  float    v8f;

constexpr int NAGENT = 8;
constexpr int NHEADS = 4;
constexpr int HYP    = 128;
constexpr int EMB    = 32;
constexpr int NE     = 16;
constexpr int ED     = 64;

// workspace regions (halves), weights pre-swizzled into B-fragment order
constexpr int SZ_FC1_H = ED * HYP;
constexpr int SZ_ATT_H = HYP * 3 * HYP;
constexpr int SZ_OUT_H = HYP * HYP;
constexpr int SZ_FC2_H = HYP * EMB;
constexpr int OFF_FC1 = 0;
constexpr int OFF_ATT = OFF_FC1 + 4 * SZ_FC1_H;
constexpr int OFF_OUT = OFF_ATT + 4 * SZ_ATT_H;
constexpr int OFF_FC2 = OFF_OUT + 4 * SZ_OUT_H;
constexpr int TOTAL_H = OFF_FC2 + 4 * SZ_FC2_H;

// per-wave LDS regions (halves) inside one 8192-half buffer
constexpr int QOFF = 0;     // x1 A-frags -> q A-frags -> w frags -> out-layer o1
constexpr int KOFF = 2048;  // kT B-frags -> attn A-frags
constexpr int VOFF = 4096;  // v B-frags (upper K half zero) -> outs f32

// ---------------- wave reductions (wave32) ----------------
__device__ __forceinline__ float wsum32(float x) {
#pragma unroll
  for (int m = 16; m > 0; m >>= 1) x += __shfl_xor(x, m, 32);
  return x;
}
__device__ __forceinline__ float wmax32(float x) {
#pragma unroll
  for (int m = 16; m > 0; m >>= 1) x = fmaxf(x, __shfl_xor(x, m, 32));
  return x;
}
__device__ __forceinline__ float hsum16(float x) {
#pragma unroll
  for (int m = 8; m > 0; m >>= 1) x += __shfl_xor(x, m, 32);
  return x;
}
__device__ __forceinline__ float hmax16(float x) {
#pragma unroll
  for (int m = 8; m > 0; m >>= 1) x = fmaxf(x, __shfl_xor(x, m, 32));
  return x;
}

// half-slot within a 512-half 16x32 A-fragment chunk for element (m, k)
__device__ __forceinline__ int a_slot(int m, int k) {
  int e = ((k >> 1) & 8) | (k & 7);
  return (m + ((k & 8) << 1)) * 16 + e;
}

// entities A fragment: lane needs two contiguous 8-float runs
__device__ __forceinline__ v16h load_a_ent(const float* __restrict__ E, int k0,
                                           int lane) {
  int m = lane & 15, hi = lane >> 4;
  const float* p0 = E + m * ED + k0 + 8 * hi;
  v8f r0 = *(const v8f*)p0;         // K = k0 + 8*hi + 0..7
  v8f r1 = *(const v8f*)(p0 + 16);  // K = k0 + 16 + 8*hi + 0..7
  v16h a;
#pragma unroll
  for (int e = 0; e < 8; ++e) {
    a[e]     = (_Float16)r0[e];
    a[8 + e] = (_Float16)r1[e];
  }
  return a;
}

__device__ __forceinline__ v8f wmma16(const v16h a, const v16h b, const v8f c) {
  return __builtin_amdgcn_wmma_f32_16x16x32_f16(false, a, false, b, (short)0, c,
                                                false, false);
}

// ---------------- prep: fp32 weights -> f16 swizzled B fragments ----
__device__ __forceinline__ void swz(const float* __restrict__ W, int K, int N,
                                    int idx, _Float16* __restrict__ dst) {
  int frag = idx >> 9, within = idx & 511;
  int l = within >> 4, e = within & 15;
  int nkc = K >> 5;
  int nt = frag / nkc, kc = frag - nt * nkc;
  int n = nt * 16 + (l & 15);
  int k = kc * 32 + ((l >> 4) << 3) + (e & 7) + ((e >> 3) << 4);
  dst[idx] = (_Float16)W[k * N + n];
}

__global__ void qmix_prep(const float* __restrict__ fc1_w,
                          const float* __restrict__ attn_w,
                          const float* __restrict__ out_w,
                          const float* __restrict__ fc2_w,
                          _Float16* __restrict__ wsp) {
  int idx = blockIdx.x * blockDim.x + threadIdx.x;
  if (idx >= TOTAL_H) return;
  if (idx < OFF_ATT) {
    int r = idx - OFF_FC1, i = r / SZ_FC1_H, j = r - i * SZ_FC1_H;
    swz(fc1_w + i * ED * HYP, ED, HYP, j, wsp + OFF_FC1 + i * SZ_FC1_H);
  } else if (idx < OFF_OUT) {
    int r = idx - OFF_ATT, i = r / SZ_ATT_H, j = r - i * SZ_ATT_H;
    swz(attn_w + i * HYP * 3 * HYP, HYP, 3 * HYP, j, wsp + OFF_ATT + i * SZ_ATT_H);
  } else if (idx < OFF_FC2) {
    int r = idx - OFF_OUT, i = r / SZ_OUT_H, j = r - i * SZ_OUT_H;
    swz(out_w + i * HYP * HYP, HYP, HYP, j, wsp + OFF_OUT + i * SZ_OUT_H);
  } else {
    int r = idx - OFF_FC2, i = r / SZ_FC2_H, j = r - i * SZ_FC2_H;
    swz(fc2_w + i * HYP * EMB, HYP, EMB, j, wsp + OFF_FC2 + i * SZ_FC2_H);
  }
}

// ---------------- main: 1 block per batch element, wave i = hypernet i --------
__global__ __launch_bounds__(128) void qmix_main(
    const float* __restrict__ agent_qs, const float* __restrict__ entities,
    const int* __restrict__ emaskg, const float* __restrict__ fc1_b,
    const float* __restrict__ out_b, const float* __restrict__ fc2_b,
    const _Float16* __restrict__ wsp, float* __restrict__ out) {
  __shared__ __align__(32) _Float16 bufB[4][8192];  // 64 KB total

  const int b    = blockIdx.x;
  const int wave = threadIdx.x >> 5;
  const int lane = threadIdx.x & 31;
  const int i    = wave;
  const int l15  = lane & 15;
  const int hi_d = lane >> 4;

  const v16h* Bfc1 = (const v16h*)(wsp + OFF_FC1 + i * SZ_FC1_H);
  const v16h* Batt = (const v16h*)(wsp + OFF_ATT + i * SZ_ATT_H);
  const v16h* Bout = (const v16h*)(wsp + OFF_OUT + i * SZ_OUT_H);
  const v16h* Bfc2 = (const v16h*)(wsp + OFF_FC2 + i * SZ_FC2_H);

  _Float16* qb = bufB[i] + QOFF;
  _Float16* kb = bufB[i] + KOFF;
  _Float16* vb = bufB[i] + VOFF;

  const int maskv = emaskg[b * NE + l15];

  // per-row (r -> m = r + 8*hi) masks, computed once
  int   am8[8];
  float keep8[8];
#pragma unroll
  for (int r = 0; r < 8; ++r) {
    int m  = r + (hi_d << 3);
    int am = __shfl(maskv, m, 32);
    am8[r]   = am;
    keep8[r] = (m < NAGENT) ? (1.f - (float)am) : 0.f;
  }

  v16h zv;
#pragma unroll
  for (int e = 0; e < 16; ++e) zv[e] = (_Float16)0.f;

  // zero v region (so B-fragment K=16..31 rows are 0)
#pragma unroll
  for (int t = 0; t < 8; ++t) ((v16h*)vb)[t * 32 + lane] = zv;

  // ---- fc1: x1 = relu(E @ W1 + b1) -> A-frags in q region ----
  const float* E = entities + (size_t)b * NE * ED;
  v16h aE0 = load_a_ent(E, 0, lane);
  v16h aE1 = load_a_ent(E, 32, lane);
  for (int nt = 0; nt < 8; ++nt) {
    v8f c = {};
    c = wmma16(aE0, Bfc1[(nt * 2 + 0) * 32 + lane], c);
    c = wmma16(aE1, Bfc1[(nt * 2 + 1) * 32 + lane], c);
    int n = nt * 16 + l15;
    float bias = fc1_b[i * HYP + n];
    int sb = (n >> 5) * 512 + a_slot(0, n & 31);
#pragma unroll
    for (int r = 0; r < 8; ++r) {
      float val = c[r] + bias;
      qb[sb + (r + (hi_d << 3)) * 16] = (_Float16)(val > 0.f ? val : 0.f);
    }
  }

  // ---- qkv = x1 @ Wattn: q->A-frags (q region), kT/v->B-frags ----
  v16h aX[4];
#pragma unroll
  for (int kc = 0; kc < 4; ++kc) aX[kc] = ((const v16h*)qb)[kc * 32 + lane];
  for (int nt = 0; nt < 24; ++nt) {
    v8f c = {};
#pragma unroll
    for (int kc = 0; kc < 4; ++kc)
      c = wmma16(aX[kc], Batt[(nt * 4 + kc) * 32 + lane], c);
    int n = nt * 16 + l15;
    if (n < HYP) {                       // q: A-frag layout, agent rows only
      if (hi_d == 0) {
        int sb = (n >> 5) * 512 + a_slot(0, n & 31);
#pragma unroll
        for (int r = 0; r < 8; ++r) qb[sb + r * 16] = (_Float16)c[r];
      }
    } else if (n < 2 * HYP) {            // kT: B-frag layout (K=d, N=entity)
      int ck = n - HYP;
      int hh = ck >> 5, d = ck & 31;
      int base = hh * 512 + ((d & 8) << 5) + ((d & 16) >> 1) + (d & 7);
#pragma unroll
      for (int r = 0; r < 8; ++r)
        kb[base + (r + (hi_d << 3)) * 16] = (_Float16)c[r];
    } else {                             // v: B-frag layout (K=entity, N=d)
      int cv = n - 2 * HYP;
      int hh = cv >> 5, tt = (cv >> 4) & 1;
      v8h pk;
#pragma unroll
      for (int r = 0; r < 8; ++r) pk[r] = (_Float16)c[r];
      *(v8h*)(vb + (hh * 2 + tt) * 512 + l15 * 16 + (hi_d << 8)) = pk;
    }
  }

  // ---- attention: per head, 1 WMMA logits + reg softmax + 2 WMMA attn ----
  const float rscale = 0.17677669529663688f;  // 1/sqrt(32)
  for (int h = 0; h < NHEADS; ++h) {
    v16h qa  = ((const v16h*)qb)[h * 32 + lane];
    v16h kfr = ((const v16h*)kb)[h * 32 + lane];
    v8f lg8 = {};
    lg8 = wmma16(qa, kfr, lg8);  // D: m=agent(r+8*hi), n=entity(l15)
    // masked softmax over entities (16 lanes), write w into q chunk h
#pragma unroll
    for (int r = 0; r < 8; ++r) {
      int a   = r + (hi_d << 3);
      int pre = (am8[r] | maskv);
      float lg = pre ? -1e30f : lg8[r] * rscale;
      float mx = hmax16(lg);
      float ex = __expf(lg - mx);
      float s  = hsum16(ex);
      float w  = pre ? 0.f : (ex / s);
      qb[h * 512 + a_slot(a, l15)] = (_Float16)w;  // K = entity index
    }
    // zero attn dest chunk (k chunk h, already consumed)
    ((v16h*)kb)[h * 32 + lane] = zv;
    v16h wa = ((const v16h*)qb)[h * 32 + lane];
#pragma unroll
    for (int t = 0; t < 2; ++t) {
      v8f d8 = {};
      d8 = wmma16(wa, ((const v16h*)vb)[(h * 2 + t) * 32 + lane], d8);
      if (hi_d == 0) {  // store agent rows 0..7; pad rows stay zero
        int sb = h * 512 + a_slot(0, t * 16 + l15);
#pragma unroll
        for (int r = 0; r < 8; ++r) kb[sb + r * 16] = (_Float16)d8[r];
      }
    }
  }

  // ---- out layer: attn @ Wout + out_b, * keep -> q region (A-frags) ----
#pragma unroll
  for (int kc = 0; kc < 4; ++kc) aX[kc] = ((const v16h*)kb)[kc * 32 + lane];
  for (int nt = 0; nt < 8; ++nt) {
    v8f c = {};
#pragma unroll
    for (int kc = 0; kc < 4; ++kc)
      c = wmma16(aX[kc], Bout[(nt * 4 + kc) * 32 + lane], c);
    int n = nt * 16 + l15;
    float bias = out_b[i * HYP + n];
    int sb = (n >> 5) * 512 + a_slot(0, n & 31);
#pragma unroll
    for (int r = 0; r < 8; ++r)
      qb[sb + (r + (hi_d << 3)) * 16] = (_Float16)((c[r] + bias) * keep8[r]);
  }

  // ---- fc2 -> outs_i (8x32 f32 in v region; pad rows write zeros) ----
#pragma unroll
  for (int kc = 0; kc < 4; ++kc) aX[kc] = ((const v16h*)qb)[kc * 32 + lane];
  float* outsF = (float*)vb;
  for (int nt = 0; nt < 2; ++nt) {
    v8f c = {};
#pragma unroll
    for (int kc = 0; kc < 4; ++kc)
      c = wmma16(aX[kc], Bfc2[(nt * 4 + kc) * 32 + lane], c);
    int n = nt * 16 + l15;
    float bias = fc2_b[i * EMB + n];
#pragma unroll
    for (int r = 0; r < 8; ++r) {
      int m = r + (hi_d << 3);
      outsF[m * EMB + n] = (c[r] + bias) * keep8[r];  // rows 8..15 unused
    }
  }

  __syncthreads();

  // ---- mixing (wave 0, lane = embedding index e) ----
  if (wave == 0) {
    const float* f0 = (const float*)(bufB[0] + VOFF);
    const float* f1 = (const float*)(bufB[1] + VOFF);
    const float* f2 = (const float*)(bufB[2] + VOFF);
    const float* f3 = (const float*)(bufB[3] + VOFF);
    int e = lane;
    float hid = 0.f, b1 = 0.f, t1 = 0.f, vs = 0.f;
#pragma unroll
    for (int a = 0; a < 8; ++a) {
      float r0 = f0[a * EMB + e];
      float mx = wmax32(r0);
      float ex = __expf(r0 - mx);
      float s  = wsum32(ex);
      hid += agent_qs[b * NAGENT + a] * (ex / s);
      b1 += f2[a * EMB + e];
      t1 += f1[a * EMB + e];
      vs += f3[a * EMB + e];
    }
    b1 *= 0.125f;
    t1 *= 0.125f;
    float mx = wmax32(t1);
    float ex = __expf(t1 - mx);
    float wf = ex / wsum32(ex);
    float vmean = wsum32(vs) * (1.f / 256.f);
    float xh = hid + b1;
    float hdn = xh > 0.f ? xh : (__expf(xh) - 1.f);
    float y = wsum32(hdn * wf) + vmean;
    if (lane == 0) out[b] = y;
  }
}

extern "C" void kernel_launch(void* const* d_in, const int* in_sizes, int n_in,
                              void* d_out, int out_size, void* d_ws, size_t ws_size,
                              hipStream_t stream) {
  const float* agent_qs = (const float*)d_in[0];
  const float* entities = (const float*)d_in[1];
  const int*   emask    = (const int*)d_in[2];
  const float* fc1_w    = (const float*)d_in[3];
  const float* fc1_b    = (const float*)d_in[4];
  const float* attn_w   = (const float*)d_in[5];
  const float* out_w    = (const float*)d_in[6];
  const float* out_b    = (const float*)d_in[7];
  const float* fc2_w    = (const float*)d_in[8];
  const float* fc2_b    = (const float*)d_in[9];

  const int B = in_sizes[0] / NAGENT;  // 8192
  _Float16* wsp = (_Float16*)d_ws;

  qmix_prep<<<(TOTAL_H + 255) / 256, 256, 0, stream>>>(fc1_w, attn_w, out_w, fc2_w, wsp);
  qmix_main<<<B, 128, 0, stream>>>(agent_qs, entities, emask, fc1_b, out_b, fc2_b,
                                   wsp, (float*)d_out);
}
